// DiagnosticRNN_59605556134303
// MI455X (gfx1250) — matrix operand order repaired and checked by reading, not compiled, not run on hardware
//
#include <hip/hip_runtime.h>
#include <hip/hip_bf16.h>

typedef __attribute__((ext_vector_type(16))) __bf16 v16bf;
typedef __attribute__((ext_vector_type(8)))  __bf16 v8bf;
typedef __attribute__((ext_vector_type(8)))  float  v8f;

#define B_   2048
#define T_   128
#define V_   25
#define VP   32       // V padded to one bf16-WMMA K step
#define H_   512
#define G4   2048     // 4*H
#define NC   100
#define NCP  112      // NC padded to 16-col WMMA tiles

// ---------------------------------------------------------------------------
// Prep: convert weights to bf16 (pad W_ih K: 25->32, pad W_fc rows 100->112),
// fold biases.
// ---------------------------------------------------------------------------
__global__ void prep_weights(const float* __restrict__ Wih,
                             const float* __restrict__ Whh,
                             const float* __restrict__ bih,
                             const float* __restrict__ bhh,
                             const float* __restrict__ Wfc,
                             __bf16* __restrict__ Wx,    // [G4][VP]
                             __bf16* __restrict__ Wh,    // [G4][H]
                             __bf16* __restrict__ Wfcb,  // [NCP][H]
                             float*  __restrict__ bias)  // [G4]
{
    int idx = blockIdx.x * blockDim.x + threadIdx.x;
    int stride = gridDim.x * blockDim.x;
    for (int i = idx; i < G4 * H_; i += stride)
        Wh[i] = (__bf16)Whh[i];
    for (int i = idx; i < G4 * VP; i += stride) {
        int r = i / VP, k = i % VP;
        Wx[i] = (k < V_) ? (__bf16)Wih[r * V_ + k] : (__bf16)0.0f;
    }
    for (int i = idx; i < NCP * H_; i += stride) {
        int r = i / H_, k = i % H_;
        Wfcb[i] = (r < NC) ? (__bf16)Wfc[r * H_ + k] : (__bf16)0.0f;
    }
    for (int i = idx; i < G4; i += stride)
        bias[i] = bih[i] + bhh[i];
}

// Prep: messages f32 [B][T][25] -> bf16 [B][T][32] (zero-padded K).
__global__ void prep_x(const float* __restrict__ msg, __bf16* __restrict__ Xbf)
{
    long idx = (long)blockIdx.x * blockDim.x + threadIdx.x;
    long stride = (long)gridDim.x * blockDim.x;
    const long n = (long)B_ * T_ * VP;
    for (long i = idx; i < n; i += stride) {
        long bt = i / VP; int k = (int)(i % VP);
        Xbf[i] = (k < V_) ? (__bf16)msg[bt * V_ + k] : (__bf16)0.0f;
    }
}

// ---------------------------------------------------------------------------
// One LSTM timestep. Block = 128 threads = 4 waves; wave g computes the 16x16
// tile of gate g (i,f,g,o) for batch tile blockIdx.x, hidden tile blockIdx.y:
//   gates = X_t @ Wx^T  (1 bf16 WMMA, K padded 25->32)
//         + h   @ Wh^T  (16 bf16 WMMAs, K = 512)
// h is stored in bf16 (matmul precision) so A-fragments are direct b128
// loads with no conversion in the hot loop. c stays f32, updated in place
// (block-private region); h is ping-ponged (hIn read by all hidden tiles).
// ---------------------------------------------------------------------------
__device__ __forceinline__ float sigmoid_f(float x) {
    return 1.0f / (1.0f + __expf(-x));
}

__global__ __launch_bounds__(128)
void lstm_step(const __bf16* __restrict__ Xbf,  // [B][T][VP] bf16
               const __bf16* __restrict__ Wx,   // [G4][VP]   bf16
               const __bf16* __restrict__ Wh,   // [G4][H]    bf16
               const float*  __restrict__ bias, // [G4]
               const __bf16* __restrict__ hIn,  // [B][H] bf16
               __bf16* __restrict__ hOut,       // [B][H] bf16
               float* __restrict__ c,           // [B][H] f32 (in place)
               int t, int first)
{
    __shared__ float gact[4][16][16];

    const int g    = threadIdx.x >> 5;     // wave id = gate id (i,f,g,o)
    const int lane = threadIdx.x & 31;
    const int m    = lane & 15;            // A-frag row / B-frag column
    const int kh   = lane >> 4;            // K half select
    const int bb   = blockIdx.x * 16;      // batch tile base
    const int hh   = blockIdx.y * 16;      // hidden tile base
    const int n    = m;                    // output column within tile

    v8f acc = {0.f, 0.f, 0.f, 0.f, 0.f, 0.f, 0.f, 0.f};

    // ---- input projection: A = X_t tile (16x32 bf16), B = Wx rows ----
    {
        const __bf16* ap = Xbf + ((long)(bb + m) * T_ + t) * VP;
        v8bf lo = *(const v8bf*)(ap + kh * 8);
        v8bf hi = *(const v8bf*)(ap + 16 + kh * 8);
        v16bf a;
        #pragma unroll
        for (int j = 0; j < 8; ++j) { a[j] = lo[j]; a[8 + j] = hi[j]; }
        const __bf16* bp = Wx + (long)(g * H_ + hh + n) * VP;
        v8bf blo = *(const v8bf*)(bp + kh * 8);
        v8bf bhi = *(const v8bf*)(bp + 16 + kh * 8);
        v16bf b;
        #pragma unroll
        for (int j = 0; j < 8; ++j) { b[j] = blo[j]; b[8 + j] = bhi[j]; }
        acc = __builtin_amdgcn_wmma_f32_16x16x32_bf16(
                  false, a, false, b, (short)0, acc, false, false);
    }

    // ---- recurrent GEMM: gates += h @ Wh^T over K = 512 ----
    if (!first) {
        const __bf16* hrow = hIn + (long)(bb + m) * H_;
        const __bf16* wrow = Wh + (long)(g * H_ + hh + n) * H_;
        #pragma unroll 4
        for (int kc = 0; kc < H_; kc += 32) {
            __builtin_prefetch(wrow + kc + 128, 0, 3);   // near-scope prefetch
            v8bf lo = *(const v8bf*)(hrow + kc + kh * 8);
            v8bf hi = *(const v8bf*)(hrow + kc + 16 + kh * 8);
            v16bf a;
            #pragma unroll
            for (int j = 0; j < 8; ++j) { a[j] = lo[j]; a[8 + j] = hi[j]; }
            v8bf blo = *(const v8bf*)(wrow + kc + kh * 8);
            v8bf bhi = *(const v8bf*)(wrow + kc + 16 + kh * 8);
            v16bf b;
            #pragma unroll
            for (int j = 0; j < 8; ++j) { b[j] = blo[j]; b[8 + j] = bhi[j]; }
            acc = __builtin_amdgcn_wmma_f32_16x16x32_bf16(
                      false, a, false, b, (short)0, acc, false, false);
        }
    }

    // ---- bias + nonlinearity, publish gate tile to LDS ----
    const float bv = bias[g * H_ + hh + n];
    #pragma unroll
    for (int r = 0; r < 8; ++r) {
        float x = acc[r] + bv;                  // acc[r] = row (r + 8*kh), col n
        float y = (g == 2) ? tanhf(x) : sigmoid_f(x);
        gact[g][r + 8 * kh][n] = y;
    }
    __syncthreads();

    // ---- LSTM cell update: 256 elements over 128 threads ----
    for (int e = threadIdx.x; e < 256; e += 128) {
        int mm = e >> 4, nn = e & 15;
        float iv = gact[0][mm][nn];
        float fv = gact[1][mm][nn];
        float gv = gact[2][mm][nn];
        float ov = gact[3][mm][nn];
        long  ci = (long)(bb + mm) * H_ + hh + nn;
        float cold = first ? 0.0f : c[ci];
        float cn = fv * cold + iv * gv;
        c[ci]    = cn;
        hOut[ci] = (__bf16)(ov * tanhf(cn));
    }
}

// ---------------------------------------------------------------------------
// Final FC: out = h_last @ W_fc^T + b_fc; W_fc pre-padded to 112 bf16 rows so
// the K loop is branch-free (store-side bounds check only). 1 wave / block.
// ---------------------------------------------------------------------------
__global__ __launch_bounds__(32)
void fc_wmma(const __bf16* __restrict__ h,     // [B][H] bf16
             const __bf16* __restrict__ Wfcb,  // [NCP][H] bf16 (zero padded)
             const float*  __restrict__ bfc,   // [NC]
             float* __restrict__ out)          // [B][NC]
{
    const int lane = threadIdx.x & 31;
    const int m    = lane & 15;
    const int kh   = lane >> 4;
    const int bb   = blockIdx.x * 16;
    const int nn   = blockIdx.y * 16;
    const int col  = nn + m;                 // class index (B-frag column)

    v8f acc = {0.f, 0.f, 0.f, 0.f, 0.f, 0.f, 0.f, 0.f};
    const __bf16* hrow = h + (long)(bb + m) * H_;
    const __bf16* wrow = Wfcb + (long)col * H_;

    for (int kc = 0; kc < H_; kc += 32) {
        v8bf lo = *(const v8bf*)(hrow + kc + kh * 8);
        v8bf hi = *(const v8bf*)(hrow + kc + 16 + kh * 8);
        v16bf a;
        #pragma unroll
        for (int j = 0; j < 8; ++j) { a[j] = lo[j]; a[8 + j] = hi[j]; }
        v8bf blo = *(const v8bf*)(wrow + kc + kh * 8);
        v8bf bhi = *(const v8bf*)(wrow + kc + 16 + kh * 8);
        v16bf b;
        #pragma unroll
        for (int j = 0; j < 8; ++j) { b[j] = blo[j]; b[8 + j] = bhi[j]; }
        acc = __builtin_amdgcn_wmma_f32_16x16x32_bf16(
                  false, a, false, b, (short)0, acc, false, false);
    }

    #pragma unroll
    for (int r = 0; r < 8; ++r) {
        int row = bb + r + 8 * kh;
        if (col < NC)
            out[(long)row * NC + col] = acc[r] + bfc[col];
    }
}

// ---------------------------------------------------------------------------
// Host-side orchestration (graph-capture safe: only kernel launches on stream)
// ---------------------------------------------------------------------------
extern "C" void kernel_launch(void* const* d_in, const int* in_sizes, int n_in,
                              void* d_out, int out_size, void* d_ws, size_t ws_size,
                              hipStream_t stream)
{
    (void)in_sizes; (void)n_in; (void)out_size; (void)ws_size;

    const float* msg = (const float*)d_in[0];  // [B,T,V]
    const float* Wih = (const float*)d_in[1];  // [4H,V]
    const float* Whh = (const float*)d_in[2];  // [4H,H]
    const float* bih = (const float*)d_in[3];  // [4H]
    const float* bhh = (const float*)d_in[4];  // [4H]
    const float* Wfc = (const float*)d_in[5];  // [NC,H]
    const float* bfc = (const float*)d_in[6];  // [NC]
    float* out = (float*)d_out;                // [B,NC]

    // Workspace carve-up (~27 MB total)
    char* p = (char*)d_ws;
    __bf16* hA   = (__bf16*)p; p += (size_t)B_ * H_ * sizeof(__bf16);
    __bf16* hB   = (__bf16*)p; p += (size_t)B_ * H_ * sizeof(__bf16);
    float*  cbuf = (float*)p;  p += (size_t)B_ * H_ * sizeof(float);
    __bf16* Wh   = (__bf16*)p; p += (size_t)G4 * H_ * sizeof(__bf16);
    __bf16* Wx   = (__bf16*)p; p += (size_t)G4 * VP * sizeof(__bf16);
    __bf16* Wfcb = (__bf16*)p; p += (size_t)NCP * H_ * sizeof(__bf16);
    float*  bias = (float*)p;  p += (size_t)G4 * sizeof(float);
    __bf16* Xbf  = (__bf16*)p; p += (size_t)B_ * T_ * VP * sizeof(__bf16);

    prep_weights<<<1024, 256, 0, stream>>>(Wih, Whh, bih, bhh, Wfc,
                                           Wx, Wh, Wfcb, bias);
    prep_x<<<4096, 256, 0, stream>>>(msg, Xbf);

    dim3 sgrid(B_ / 16, H_ / 16);              // 128 x 32 blocks per step
    __bf16* hlast = hA;
    for (int t = 0; t < T_; ++t) {
        __bf16* hout = (t & 1) ? hB : hA;
        __bf16* hin  = (t & 1) ? hA : hB;      // garbage at t==0; skipped (first)
        lstm_step<<<sgrid, 128, 0, stream>>>(Xbf, Wx, Wh, bias,
                                             hin, hout, cbuf, t, (t == 0) ? 1 : 0);
        hlast = hout;
    }

    fc_wmma<<<dim3(B_ / 16, (NC + 15) / 16), 32, 0, stream>>>(hlast, Wfcb, bfc, out);
}
